// LongNetTransformer_4990751998415
// MI455X (gfx1250) — matrix operand order repaired and checked by reading, not compiled, not run on hardware
//
#include <hip/hip_runtime.h>
#include <hip/hip_bf16.h>

// ---------------------------------------------------------------------------
// LongNet dilated attention, 3 layers, on gfx1250 via bf16 WMMA (16x16x32).
// GEMMs: 32x64 per-wave tiles, K unrolled x2 with ping-pong fragment buffers
// (no cross-buffer register copies; loads overlap the matrix pipe).
// ---------------------------------------------------------------------------

typedef __bf16 bf16_t;
typedef __attribute__((ext_vector_type(8)))  bf16_t v8bf;
typedef __attribute__((ext_vector_type(16))) bf16_t v16bf;
typedef __attribute__((ext_vector_type(8)))  float  v8f;
typedef __attribute__((ext_vector_type(4)))  float  v4f;

#define D_MODEL 1024
#define H_HEADS 16
#define HD      64
#define BATCH   2
#define SEQLEN  8192
#define M_ROWS  8192   // B * S/dr (dr == 2 for all layers)

__device__ __forceinline__ unsigned short f2bf(float f) {
  union { float f; unsigned u; } v; v.f = f;
  unsigned u = v.u;
  u += 0x7FFFu + ((u >> 16) & 1u);       // round-to-nearest-even
  return (unsigned short)(u >> 16);
}

__device__ __forceinline__ v16bf cat8(v8bf a, v8bf b) {
  return __builtin_shufflevector(a, b, 0,1,2,3,4,5,6,7,8,9,10,11,12,13,14,15);
}

// A-fragment (16x32 bf16): lane<16 row=lane holds K 0..15; lane>=16
// holds K 16..31.  rowk0 = &A[row*ld + k0].
__device__ __forceinline__ v16bf ldA(const bf16_t* rowk0, int lane) {
  int kh = (lane >> 4) << 4;
  v8bf lo = *(const v8bf*)(rowk0 + kh);
  v8bf hi = *(const v8bf*)(rowk0 + kh + 8);
  return cat8(lo, hi);
}

// B-fragment (32x16 bf16): lane = column, lane<16 -> K 0..15, lane>=16 -> K
// 16..31; p points at the lane's 16 contiguous K values.
__device__ __forceinline__ v16bf ldB16(const bf16_t* p) {
  v8bf lo = *(const v8bf*)p;
  v8bf hi = *(const v8bf*)(p + 8);
  return cat8(lo, hi);
}

__device__ __forceinline__ v8f wmma_bf16(v16bf a, v16bf b, v8f c) {
  return __builtin_amdgcn_wmma_f32_16x16x32_bf16(false, a, false, b,
                                                 (short)0, c, false, false);
}

// ---------------------------------------------------------------------------
// Shared 32x64-tile GEMM K-loop (K = 1024 = 32 steps of 32), ping-pong
// double buffering, no register copies.  NT16 = N/16 (weight tile stride).
template <int NT16>
__device__ __forceinline__ void gemm_kloop(const bf16_t* A0, const bf16_t* A1,
                                           const bf16_t* Wp, int lane,
                                           v8f acc0[4], v8f acc1[4]) {
  const long WS = (long)NT16 * 512;      // elements per K-step
  v16bf a00, a10, b0[4], a01, a11, b1[4];

  a00 = ldA(A0, lane);
  a10 = ldA(A1, lane);
#pragma unroll
  for (int i = 0; i < 4; ++i) b0[i] = ldB16(Wp + i * 512);

  for (int it = 0; it < 15; ++it) {
    // load set1 (k = 2it+1)
    a01 = ldA(A0 + 32, lane);
    a11 = ldA(A1 + 32, lane);
#pragma unroll
    for (int i = 0; i < 4; ++i) b1[i] = ldB16(Wp + WS + i * 512);
    // compute set0 (k = 2it)
#pragma unroll
    for (int i = 0; i < 4; ++i) {
      acc0[i] = wmma_bf16(a00, b0[i], acc0[i]);
      acc1[i] = wmma_bf16(a10, b0[i], acc1[i]);
    }
    A0 += 64; A1 += 64; Wp += 2 * WS;
    // load set0 (k = 2it+2)
    a00 = ldA(A0, lane);
    a10 = ldA(A1, lane);
#pragma unroll
    for (int i = 0; i < 4; ++i) b0[i] = ldB16(Wp + i * 512);
    // compute set1 (k = 2it+1)
#pragma unroll
    for (int i = 0; i < 4; ++i) {
      acc0[i] = wmma_bf16(a01, b1[i], acc0[i]);
      acc1[i] = wmma_bf16(a11, b1[i], acc1[i]);
    }
  }
  // epilogue: k = 30 (set0), k = 31 (set1)
  a01 = ldA(A0 + 32, lane);
  a11 = ldA(A1 + 32, lane);
#pragma unroll
  for (int i = 0; i < 4; ++i) b1[i] = ldB16(Wp + WS + i * 512);
#pragma unroll
  for (int i = 0; i < 4; ++i) {
    acc0[i] = wmma_bf16(a00, b0[i], acc0[i]);
    acc1[i] = wmma_bf16(a10, b0[i], acc1[i]);
  }
#pragma unroll
  for (int i = 0; i < 4; ++i) {
    acc0[i] = wmma_bf16(a01, b1[i], acc0[i]);
    acc1[i] = wmma_bf16(a11, b1[i], acc1[i]);
  }
}

// ---------------------------------------------------------------------------
// x fp32 -> bf16, 4 elements per thread
__global__ __launch_bounds__(256) void k_cvt_x(const float* __restrict__ x,
                                               unsigned short* __restrict__ y,
                                               long n4) {
  long i = (long)blockIdx.x * 256 + threadIdx.x;
  if (i >= n4) return;
  v4f f = *(const v4f*)(x + i * 4);
  union { unsigned short h[4]; uint2 u; } o;
  o.h[0] = f2bf(f.x); o.h[1] = f2bf(f.y);
  o.h[2] = f2bf(f.z); o.h[3] = f2bf(f.w);
  *(uint2*)(y + i * 4) = o.u;
}

// ---------------------------------------------------------------------------
// Swizzle fp32 weight W[K=1024, N] (row-major) into WMMA B-fragment tiles:
// out[((kt*(N/16)+nt)*32 + lane)*16 + t] = bf16(W[(kt*32+(lane>>4)*16+t)*N + nt*16+(lane&15)])
__global__ __launch_bounds__(256) void k_swizzle_w(const float* __restrict__ W,
                                                   unsigned short* __restrict__ out,
                                                   int N) {
  long idx = (long)blockIdx.x * 256 + threadIdx.x;
  long total = (long)1024 * N;
  if (idx >= total) return;
  int t    = (int)(idx & 15);
  int l    = (int)((idx >> 4) & 31);
  long tile = idx >> 9;
  int nt16 = N >> 4;
  int nt = (int)(tile % nt16);
  int kt = (int)(tile / nt16);
  int col = nt * 16 + (l & 15);
  int kk  = kt * 32 + ((l >> 4) << 4) + t;
  out[idx] = f2bf(W[(long)kk * N + col]);
}

// ---------------------------------------------------------------------------
// QKV projection (M=8192, K=1024, N=3072) with the dilated gather folded into
// the A addressing.  Each wave: 32x64 output tile.
// Writes Q,K row-major bf16 [8192,1024]; V transposed [(segid*16+h)*64+d][s].
__global__ __launch_bounds__(256) void k_qkv_gemm(
    const unsigned short* __restrict__ xbf,
    const unsigned short* __restrict__ wsw,
    const float* __restrict__ bias,      // [3072]
    unsigned short* __restrict__ Qb,
    unsigned short* __restrict__ Kb,
    unsigned short* __restrict__ Vt,
    int seg, int s, int Nseg) {
  int wave = threadIdx.x >> 5;
  int lane = threadIdx.x & 31;
  int mtile = blockIdx.y * 8 + wave;    // 0..255, 32 rows each
  int n64   = blockIdx.x;               // 0..47
  long g0 = (long)mtile * 32;

  // tile-uniform segment decomposition (32-row tiles never cross a segment)
  int b  = (int)(g0 >> 12);
  int p  = (int)(g0 & 4095);
  int n  = p / s;
  int j0 = p - n * s;
  long segid = (long)b * Nseg + n;

  int r = lane & 15;
  const bf16_t* A0 = (const bf16_t*)xbf +
      ((long)b * SEQLEN + (long)n * seg + 2 * (j0 + r)) * D_MODEL;
  const bf16_t* A1 = A0 + 32 * D_MODEL;              // rows j0+16..j0+31
  const bf16_t* Wp = (const bf16_t*)wsw + (long)n64 * 2048 + lane * 16;

  v8f acc0[4] = {}, acc1[4] = {};
  gemm_kloop<192>(A0, A1, Wp, lane, acc0, acc1);

  int mr = (lane >> 4) << 3;
  int cl = lane & 15;
#pragma unroll
  for (int half = 0; half < 2; ++half) {
    v8f* acc = half ? acc1 : acc0;
    long gh = g0 + half * 16;
    int jh = j0 + half * 16;
#pragma unroll
    for (int i = 0; i < 4; ++i) {
      int c = n64 * 64 + i * 16 + cl;
      float bb = bias[c];
#pragma unroll
      for (int v = 0; v < 8; ++v) {
        int m = mr + v;
        long gg = gh + m;
        unsigned short hv = f2bf(acc[i][v] + bb);
        if (c < 1024) {
          Qb[gg * D_MODEL + c] = hv;
        } else if (c < 2048) {
          Kb[gg * D_MODEL + (c - 1024)] = hv;
        } else {
          int cc = c - 2048;
          int hh = cc >> 6;
          int dd = cc & 63;
          Vt[(((segid * H_HEADS + hh) << 6) + dd) * (long)s + (jh + m)] = hv;
        }
      }
    }
  }
}

// ---------------------------------------------------------------------------
// Attention within a dilated segment: one workgroup per (segment, head).
// scores = QK^T/8 (WMMA) -> fp32 LDS -> softmax -> bf16 LDS -> attn*V (WMMA).
__global__ __launch_bounds__(256) void k_attn(
    const unsigned short* __restrict__ Qb,
    const unsigned short* __restrict__ Kb,
    const unsigned short* __restrict__ Vt,
    unsigned short* __restrict__ Ob,
    int s, int Nseg) {
  __shared__ __align__(16) float          sc[128 * 132];
  __shared__ __align__(16) unsigned short at[128 * 128];

  int segh = blockIdx.x;
  int h = segh & 15;
  long segid = segh >> 4;                 // b*Nseg + n
  long bidx = segid / Nseg;
  long nidx = segid - bidx * Nseg;
  long gbase = bidx * 4096 + nidx * s;    // first gathered token of segment

  int wave = threadIdx.x >> 5;
  int lane = threadIdx.x & 31;
  int st = s >> 4;                        // 16-tiles per side
  int cl = lane & 15;

  // ---- phase 1: scores = (Q K^T) * 1/8 ----
  for (int t = wave; t < st * st; t += 8) {
    int qt = t / st, kt = t - qt * st;
    const bf16_t* qrow = (const bf16_t*)Qb +
        (gbase + qt * 16 + cl) * (long)D_MODEL + h * HD;
    const bf16_t* krow = (const bf16_t*)Kb +
        (gbase + kt * 16 + cl) * (long)D_MODEL + h * HD + ((lane >> 4) << 4);
    v8f acc = {};
#pragma unroll
    for (int ks = 0; ks < 2; ++ks) {
      v16bf a = ldA(qrow + ks * 32, lane);
      v16bf bfr = ldB16(krow + ks * 32);
      acc = wmma_bf16(a, bfr, acc);
    }
    int mr = (lane >> 4) << 3;
    int col = kt * 16 + cl;
#pragma unroll
    for (int v = 0; v < 8; ++v)
      sc[(qt * 16 + mr + v) * (s + 4) + col] = acc[v] * 0.125f;
  }
  __syncthreads();

  // ---- phase 2: row softmax, write bf16 attn ----
  if ((int)threadIdx.x < s) {
    int rb = threadIdx.x * (s + 4);
    float mx = -3.0e38f;
    for (int c = 0; c < s; ++c) mx = fmaxf(mx, sc[rb + c]);
    float sum = 0.f;
    for (int c = 0; c < s; ++c) {
      float e = __expf(sc[rb + c] - mx);
      sc[rb + c] = e;
      sum += e;
    }
    float inv = 1.f / sum;
    for (int c = 0; c < s; ++c)
      at[threadIdx.x * s + c] = f2bf(sc[rb + c] * inv);
  }
  __syncthreads();

  // ---- phase 3: O = attn @ V  (s x 64) ----
  const bf16_t* atp = (const bf16_t*)at;
  for (int t = wave; t < st * 4; t += 8) {
    int qt = t >> 2, dt = t & 3;
    const bf16_t* arow = atp + (qt * 16 + cl) * s;
    const bf16_t* vrow = (const bf16_t*)Vt +
        (((segid * H_HEADS + h) << 6) + dt * 16 + cl) * (long)s +
        ((lane >> 4) << 4);
    v8f acc = {};
    for (int ks = 0; ks < (s >> 5); ++ks) {
      v16bf a = ldA(arow + ks * 32, lane);
      v16bf bfr = ldB16(vrow + ks * 32);
      acc = wmma_bf16(a, bfr, acc);
    }
    int mr = (lane >> 4) << 3;
    int d = dt * 16 + cl;
#pragma unroll
    for (int v = 0; v < 8; ++v)
      Ob[(gbase + qt * 16 + mr + v) * (long)D_MODEL + h * HD + d] = f2bf(acc[v]);
  }
}

// ---------------------------------------------------------------------------
// out (+)= O @ Wo + bo.  32x64 tile per wave.
__global__ __launch_bounds__(256) void k_out_gemm(
    const unsigned short* __restrict__ Ob,
    const unsigned short* __restrict__ wsw,
    const float* __restrict__ bo,
    float* __restrict__ out,
    int first) {
  int wave = threadIdx.x >> 5;
  int lane = threadIdx.x & 31;
  int mtile = blockIdx.y * 8 + wave;     // 0..255, 32 rows each
  int n64   = blockIdx.x;                // 0..15
  long m0 = (long)mtile * 32;

  const bf16_t* A0 = (const bf16_t*)Ob + (m0 + (lane & 15)) * D_MODEL;
  const bf16_t* A1 = A0 + 16 * D_MODEL;
  const bf16_t* Wp = (const bf16_t*)wsw + (long)n64 * 2048 + lane * 16;

  v8f acc0[4] = {}, acc1[4] = {};
  gemm_kloop<64>(A0, A1, Wp, lane, acc0, acc1);

  int mr = (lane >> 4) << 3;
  int cl = lane & 15;
#pragma unroll
  for (int half = 0; half < 2; ++half) {
    v8f* acc = half ? acc1 : acc0;
    long mh = m0 + half * 16;
#pragma unroll
    for (int i = 0; i < 4; ++i) {
      int c = n64 * 64 + i * 16 + cl;
      float bb = bo[c];
#pragma unroll
      for (int v = 0; v < 8; ++v) {
        long idx = (mh + mr + v) * D_MODEL + c;
        float val = acc[i][v] + bb;
        out[idx] = first ? val : (out[idx] + val);
      }
    }
  }
}

// ---------------------------------------------------------------------------
extern "C" void kernel_launch(void* const* d_in, const int* in_sizes, int n_in,
                              void* d_out, int out_size, void* d_ws, size_t ws_size,
                              hipStream_t stream) {
  const float* x    = (const float*)d_in[0];   // [2, 8192, 1024]
  const float* Wqkv = (const float*)d_in[1];   // [3, 1024, 3072]
  const float* bqkv = (const float*)d_in[2];   // [3, 3072]
  const float* Wo   = (const float*)d_in[3];   // [3, 1024, 1024]
  const float* bo   = (const float*)d_in[4];   // [3, 1024]
  float* out = (float*)d_out;                  // [2, 4096, 1024]

  char* ws = (char*)d_ws;
  const size_t XBF = (size_t)BATCH * SEQLEN * D_MODEL * 2;  // 32 MiB
  const size_t WSW = (size_t)1024 * 3072 * 2;               // 6 MiB (max, reused)
  const size_t BUF = (size_t)M_ROWS * D_MODEL * 2;          // 16 MiB

  unsigned short* xbf = (unsigned short*)(ws);
  unsigned short* wsw = (unsigned short*)(ws + XBF);
  unsigned short* Qb  = (unsigned short*)(ws + XBF + WSW);
  unsigned short* Kb  = (unsigned short*)(ws + XBF + WSW + BUF);
  unsigned short* Vt  = (unsigned short*)(ws + XBF + WSW + 2 * BUF);
  unsigned short* Ob  = (unsigned short*)(ws + XBF + WSW + 3 * BUF);

  // x -> bf16 (vectorized x4)
  {
    long n4 = (long)BATCH * SEQLEN * D_MODEL / 4;
    k_cvt_x<<<dim3((unsigned)((n4 + 255) / 256)), dim3(256), 0, stream>>>(x, xbf, n4);
  }

  const int segs[3] = {64, 128, 256};
  for (int L = 0; L < 3; ++L) {
    int seg = segs[L];
    int s = seg / 2;
    int Nseg = SEQLEN / seg;

    // swizzle Wqkv[L] -> WMMA B layout
    {
      long n = (long)1024 * 3072;
      k_swizzle_w<<<dim3((unsigned)((n + 255) / 256)), dim3(256), 0, stream>>>(
          Wqkv + (long)L * 1024 * 3072, wsw, 3072);
    }
    // QKV projection with dilated gather (32x64 tiles: grid 48 x 32)
    k_qkv_gemm<<<dim3(48, 32), dim3(256), 0, stream>>>(
        xbf, wsw, bqkv + (long)L * 3072, Qb, Kb, Vt, seg, s, Nseg);

    // per-(segment, head) attention
    k_attn<<<dim3((unsigned)(BATCH * Nseg * H_HEADS)), dim3(256), 0, stream>>>(
        Qb, Kb, Vt, Ob, s, Nseg);

    // swizzle Wo[L] (wsw is free again — stream ordered)
    {
      long n = (long)1024 * 1024;
      k_swizzle_w<<<dim3((unsigned)((n + 255) / 256)), dim3(256), 0, stream>>>(
          Wo + (long)L * 1024 * 1024, wsw, 1024);
    }
    // output projection, accumulate across layers (32x64 tiles: grid 16 x 32)
    k_out_gemm<<<dim3(16, 32), dim3(256), 0, stream>>>(
        Ob, wsw, bo + (long)L * 1024, out, L == 0 ? 1 : 0);
  }
  (void)in_sizes; (void)n_in; (void)out_size; (void)ws_size;
}
